// GTN_65472481460781
// MI455X (gfx1250) — matrix operand order repaired and checked by reading, not compiled, not run on hardware
//
#include <hip/hip_runtime.h>
#include <hip/hip_bf16.h>

typedef __attribute__((ext_vector_type(16))) _Float16 v16h;
typedef __attribute__((ext_vector_type(8)))  _Float16 v8h;
typedef __attribute__((ext_vector_type(8)))  float    v8f;

#define BM 128
#define BN 64
#define BK 32

// ---------------------------------------------------------------------------
// f16 x f16 -> f32 GEMM on v_wmma_f32_16x16x32_f16 (wave32).
// 256 threads = 8 waves (4 M-waves x 2 N-waves); wave tile 32x32 = 2x2 frags.
//
// REQUIRES: M % BM == 0 and K % BK == 0 (true for every call in this model:
// M = 2048, K in {2048, 256, 128}).  N may be ragged only when FULLN=false.
//
// LDS tiles are stored pre-swizzled into the ISA fragment layouts so that
// every fragment load is one aligned 256-bit LDS read:
//   A (16x32/lane-half): lane = hs*16 + (row&15), elem e: K = (e&7)+((e>>3)<<4)+hs*8
//     -> an aligned 8-half row chunk (k multiple of 8) is CONTIGUOUS in frag space.
//   B (32x16): lane = ((k>>3)&1)*16 + (col&15), elem e = (k&7)+((k>>4)<<3).
// Epilogue (compile-time): out = maybe_relu(acc * rowScale[m] + bias[n]).
// ---------------------------------------------------------------------------
template<bool TRANSA, bool FULLN, bool HASBIAS, bool HASRS, bool RELU>
__global__ __launch_bounds__(256)
void wmma_gemm(const _Float16* __restrict__ A, const _Float16* __restrict__ B,
               float* __restrict__ C,
               int N, int K, int lda, int ldb, int ldc,
               long long sA, long long sB, long long sC,
               const float* __restrict__ bias,
               const float* __restrict__ rowScale, long long sRS)
{
    __shared__ __align__(64) _Float16 shA[BM * BK];   // 8 KB, frag-swizzled
    __shared__ __align__(64) _Float16 shB[BK * BN];   // 4 KB, frag-swizzled

    const int z = blockIdx.z;
    A += (long long)z * sA;
    B += (long long)z * sB;
    C += (long long)z * sC;
    if (HASRS) rowScale += (long long)z * sRS;

    const int tid  = threadIdx.x;
    const int lane = tid & 31;
    const int wm   = (tid >> 5) >> 1;   // 0..3
    const int wn   = (tid >> 5) & 1;    // 0..1

    const int rowBase = blockIdx.y * BM;
    const int colBase = blockIdx.x * BN;

    v8f acc[2][2];
#pragma unroll
    for (int i = 0; i < 2; ++i)
#pragma unroll
        for (int j = 0; j < 2; ++j)
#pragma unroll
            for (int e = 0; e < 8; ++e) acc[i][j][e] = 0.0f;

    for (int k0 = 0; k0 < K; k0 += BK) {
        if (k0 + BK < K) {
            __builtin_prefetch(TRANSA ? (const void*)(A + (long long)(k0 + BK) * lda + rowBase)
                                      : (const void*)(A + (long long)rowBase * lda + (k0 + BK)), 0, 1);
            __builtin_prefetch((const void*)(B + (long long)(k0 + BK) * ldb + colBase), 0, 1);
        }

        // ---- Stage A: 512 chunks of 8 halves; chunk (row r, k-quarter c4)
        //      maps to a contiguous 8-half run in fragment space. No M guard:
        //      M % BM == 0 by contract.
#pragma unroll
        for (int it = 0; it < 2; ++it) {
            const int q  = tid + it * 256;
            const int r  = q >> 2;
            const int c4 = q & 3;
            const int gr = rowBase + r;
            const int hs = c4 & 1;            // lane-half select
            const int eb = (c4 >> 1) * 8;     // element base within lane vector
            const int dst = (((r >> 4) * 32 + hs * 16 + (r & 15)) << 4) + eb;
            v8h av;
            if (TRANSA) {
#pragma unroll
                for (int j = 0; j < 8; ++j)
                    av[j] = A[(long long)(k0 + c4 * 8 + j) * lda + gr];
            } else {
                av = *(const v8h*)(A + (long long)gr * lda + (k0 + c4 * 8));
            }
            *(v8h*)(&shA[dst]) = av;
        }

        // ---- Stage B: 256 chunks of 8 halves (row k, 8 cols); vector global
        //      load, 8 x b16 scatter into fragment space.
        {
            const int k   = tid >> 3;
            const int cc0 = (tid & 7) * 8;
            const int gc0 = colBase + cc0;
            v8h bv;
            if (FULLN || gc0 + 8 <= N) {
                bv = *(const v8h*)(B + (long long)(k0 + k) * ldb + gc0);
            } else {
#pragma unroll
                for (int j = 0; j < 8; ++j)
                    bv[j] = (gc0 + j < N) ? B[(long long)(k0 + k) * ldb + gc0 + j]
                                          : (_Float16)0.0f;
            }
            const int g     = cc0 >> 4;
            const int kl    = (k & 7) + ((k >> 4) << 3);
            const int lbase = ((k >> 3) & 1) * 16 + (cc0 & 15);
#pragma unroll
            for (int j = 0; j < 8; ++j)
                shB[((g * 32 + lbase + j) << 4) + kl] = bv[j];
        }
        __syncthreads();

        // ---- Fragment loads: one aligned 256-bit LDS read each.
        v16h afrag[2], bfrag[2];
#pragma unroll
        for (int fm = 0; fm < 2; ++fm)
            afrag[fm] = *(const v16h*)(&shA[((wm * 2 + fm) * 32 + lane) << 4]);
#pragma unroll
        for (int fn = 0; fn < 2; ++fn)
            bfrag[fn] = *(const v16h*)(&shB[((wn * 2 + fn) * 32 + lane) << 4]);

#pragma unroll
        for (int fm = 0; fm < 2; ++fm)
#pragma unroll
            for (int fn = 0; fn < 2; ++fn)
                acc[fm][fn] = __builtin_amdgcn_wmma_f32_16x16x32_f16(
                    false, afrag[fm], false, bfrag[fn],
                    (short)0, acc[fm][fn], false, false);
        __syncthreads();
    }

    // ---- Store: C/D layout -> row = hs*8 + r, col = lane&15.
    const int hs = lane >> 4;
    const int lr = lane & 15;
#pragma unroll
    for (int fm = 0; fm < 2; ++fm)
#pragma unroll
        for (int fn = 0; fn < 2; ++fn) {
            const int rbase = rowBase + wm * 32 + fm * 16 + hs * 8;
            const int col   = colBase + wn * 32 + fn * 16 + lr;
            if (FULLN || col < N) {
                const float bv = HASBIAS ? bias[col] : 0.0f;
#pragma unroll
                for (int r = 0; r < 8; ++r) {
                    const int row = rbase + r;
                    float val = acc[fm][fn][r];
                    if (HASRS) val *= rowScale[row];
                    if (HASBIAS) val += bv;
                    if (RELU) val = fmaxf(val, 0.0f);
                    C[(long long)row * ldc + col] = val;
                }
            }
        }
}

// ---------------------------------------------------------------------------
// Softmax over edge types: w [5,2] -> f [5,2] along dim 0.
// ---------------------------------------------------------------------------
__global__ void softmax_ec(const float* __restrict__ w, float* __restrict__ f)
{
    const int c = threadIdx.x;
    if (c < 2) {
        float m = -1e30f;
        for (int e = 0; e < 5; ++e) m = fmaxf(m, w[e * 2 + c]);
        float ex[5], s = 0.0f;
        for (int e = 0; e < 5; ++e) { ex[e] = expf(w[e * 2 + c] - m); s += ex[e]; }
        for (int e = 0; e < 5; ++e) f[e * 2 + c] = ex[e] / s;
    }
}

// gtconv for three filter sets at once; reads A [5,N,N] once, emits f16.
__global__ void gtconv_pack(const float* __restrict__ A,
                            const float* __restrict__ F1, const float* __restrict__ F2,
                            const float* __restrict__ FB,
                            _Float16* __restrict__ P1, _Float16* __restrict__ P2,
                            _Float16* __restrict__ PB, long long NN)
{
    const long long pos = (long long)blockIdx.x * blockDim.x + threadIdx.x;
    if (pos >= NN) return;
    float a[5];
#pragma unroll
    for (int e = 0; e < 5; ++e) a[e] = A[(long long)e * NN + pos];
#pragma unroll
    for (int c = 0; c < 2; ++c) {
        float s1 = 0.f, s2 = 0.f, sb = 0.f;
#pragma unroll
        for (int e = 0; e < 5; ++e) {
            s1 += F1[e * 2 + c] * a[e];
            s2 += F2[e * 2 + c] * a[e];
            sb += FB[e * 2 + c] * a[e];
        }
        P1[(long long)c * NN + pos] = (_Float16)s1;
        P2[(long long)c * NN + pos] = (_Float16)s2;
        PB[(long long)c * NN + pos] = (_Float16)sb;
    }
}

// Column sums: deg[c*N+j] = sum_i H[c][i][j]
__global__ void colsum_k(const float* __restrict__ H, float* __restrict__ deg, int N)
{
    const int t = blockIdx.x * blockDim.x + threadIdx.x;
    if (t >= 2 * N) return;
    const int c = t / N, j = t % N;
    const float* base = H + (long long)c * N * N + j;
    float s = 0.0f;
    for (int i = 0; i < N; ++i) s += base[(long long)i * N];
    deg[t] = s;
}

// Column-scale H by dinv (deg>0 ? 1/deg : 0), write back f32 and f16 copy.
__global__ void norm_cols(float* __restrict__ H, _Float16* __restrict__ Hh,
                          const float* __restrict__ deg, int N)
{
    const long long NN = (long long)N * N;
    const long long idx = (long long)blockIdx.x * blockDim.x + threadIdx.x;
    if (idx >= 2 * NN) return;
    const int c = (int)(idx / NN);
    const int j = (int)(idx % N);
    const float d = deg[c * N + j];
    const float s = d > 0.0f ? 1.0f / d : 0.0f;
    const float v = H[idx] * s;
    H[idx]  = v;
    Hh[idx] = (_Float16)v;
}

// dout[c*N+n] = clip(rowsum, 1)^-1/2
__global__ void rowsum_dout(const float* __restrict__ H, float* __restrict__ dout, int N)
{
    const int t = blockIdx.x * blockDim.x + threadIdx.x;
    if (t >= 2 * N) return;
    const int c = t / N, n = t % N;
    const float* base = H + (long long)c * N * N + (long long)n * N;
    float s = 0.0f;
    for (int m = 0; m < N; ++m) s += base[m];
    dout[t] = rsqrtf(fmaxf(s, 1.0f));
}

// din = clip(colsum, 1)^-1/2
__global__ void din_k(const float* __restrict__ cs, float* __restrict__ din, int n)
{
    const int t = blockIdx.x * blockDim.x + threadIdx.x;
    if (t < n) din[t] = rsqrtf(fmaxf(cs[t], 1.0f));
}

// S[c][n][d] = f16(dout[c][n] * hw[n][d])
__global__ void scale_src(const float* __restrict__ hw, const float* __restrict__ dout,
                          _Float16* __restrict__ S, int N, int D)
{
    const long long idx = (long long)blockIdx.x * blockDim.x + threadIdx.x;
    const long long tot = (long long)2 * N * D;
    if (idx >= tot) return;
    const int c   = (int)(idx / ((long long)N * D));
    const long long rem = idx % ((long long)N * D);
    const int n   = (int)(rem / D);
    S[idx] = (_Float16)(dout[c * N + n] * hw[rem]);
}

__global__ void cvt_f32_f16(const float* __restrict__ in, _Float16* __restrict__ out, long long n)
{
    const long long idx = (long long)blockIdx.x * blockDim.x + threadIdx.x;
    if (idx < n) out[idx] = (_Float16)in[idx];
}

// ---------------------------------------------------------------------------
extern "C" void kernel_launch(void* const* d_in, const int* in_sizes, int n_in,
                              void* d_out, int out_size, void* d_ws, size_t ws_size,
                              hipStream_t stream)
{
    (void)in_sizes; (void)n_in; (void)out_size; (void)ws_size;
    const int N = 2048, E = 5, C = 2, WIN = 256, WOUT = 128, NCLS = 16;
    const long long NN = (long long)N * N;

    const float* A     = (const float*)d_in[0];
    const float* h     = (const float*)d_in[1];
    const float* w1    = (const float*)d_in[2];
    const float* w2    = (const float*)d_in[3];
    const float* wl    = (const float*)d_in[4];   // [1,5,2]
    const float* gcn_w = (const float*)d_in[5];
    const float* gcn_b = (const float*)d_in[6];
    const float* l1w   = (const float*)d_in[7];
    const float* l1b   = (const float*)d_in[8];
    const float* l2w   = (const float*)d_in[9];
    const float* l2b   = (const float*)d_in[10];
    float* out = (float*)d_out;

    char* wsp = (char*)d_ws;
    auto alloc = [&](size_t bytes) -> void* {
        char* p = wsp;
        wsp += (bytes + 255) & ~(size_t)255;
        return (void*)p;
    };

    float*     F1   = (float*)alloc(E * C * sizeof(float));
    float*     F2   = (float*)alloc(E * C * sizeof(float));
    float*     FB   = (float*)alloc(E * C * sizeof(float));
    _Float16*  P1   = (_Float16*)alloc((size_t)C * NN * 2);
    _Float16*  P2   = (_Float16*)alloc((size_t)C * NN * 2);
    _Float16*  PB   = (_Float16*)alloc((size_t)C * NN * 2);
    float*     H    = (float*)alloc((size_t)C * NN * 4);
    _Float16*  Hh   = (_Float16*)alloc((size_t)C * NN * 2);
    float*     deg  = (float*)alloc((size_t)C * N * 4);
    float*     deg2 = (float*)alloc((size_t)C * N * 4);
    float*     dout = (float*)alloc((size_t)C * N * 4);
    float*     din  = (float*)alloc((size_t)C * N * 4);
    _Float16*  hh   = (_Float16*)alloc((size_t)N * WIN * 2);
    _Float16*  gw16 = (_Float16*)alloc((size_t)WIN * WOUT * 2);
    float*     hw   = (float*)alloc((size_t)N * WOUT * 4);
    _Float16*  S    = (_Float16*)alloc((size_t)C * N * WOUT * 2);
    float*     X    = (float*)alloc((size_t)N * C * WOUT * 4);
    _Float16*  Xh   = (_Float16*)alloc((size_t)N * C * WOUT * 2);
    _Float16*  l1wh = (_Float16*)alloc((size_t)(C * WOUT) * WOUT * 2);
    float*     X1   = (float*)alloc((size_t)N * WOUT * 4);
    _Float16*  X1h  = (_Float16*)alloc((size_t)N * WOUT * 2);
    _Float16*  l2wh = (_Float16*)alloc((size_t)WOUT * NCLS * 2);

    // 1) softmax filters
    softmax_ec<<<1, 32, 0, stream>>>(w1, F1);
    softmax_ec<<<1, 32, 0, stream>>>(w2, F2);
    softmax_ec<<<1, 32, 0, stream>>>(wl, FB);

    // 2) gtconv -> f16 (single pass over A)
    gtconv_pack<<<(int)((NN + 255) / 256), 256, 0, stream>>>(A, F1, F2, FB, P1, P2, PB, NN);

    // 3) Layer 0: H[c] = P1[c] @ P2[c]   (batched 2048^3 f16 WMMA)
    dim3 gsq((N + BN - 1) / BN, (N + BM - 1) / BM, C);
    wmma_gemm<false,true,false,false,false><<<gsq, 256, 0, stream>>>(
        P1, P2, H, N, N, N, N, N, NN, NN, NN, nullptr, nullptr, 0);
    colsum_k<<<(2 * N + 255) / 256, 256, 0, stream>>>(H, deg, N);
    norm_cols<<<(int)((2 * NN + 255) / 256), 256, 0, stream>>>(H, Hh, deg, N);

    // 4) Layer 1: H[c] = H[c] @ B[c]
    wmma_gemm<false,true,false,false,false><<<gsq, 256, 0, stream>>>(
        Hh, PB, H, N, N, N, N, N, NN, NN, NN, nullptr, nullptr, 0);
    colsum_k<<<(2 * N + 255) / 256, 256, 0, stream>>>(H, deg, N);
    norm_cols<<<(int)((2 * NN + 255) / 256), 256, 0, stream>>>(H, Hh, deg, N);

    // 5) GCN norms on final H
    rowsum_dout<<<(2 * N + 255) / 256, 256, 0, stream>>>(H, dout, N);
    colsum_k<<<(2 * N + 255) / 256, 256, 0, stream>>>(H, deg2, N);
    din_k<<<(2 * N + 255) / 256, 256, 0, stream>>>(deg2, din, 2 * N);

    // 6) hw = h @ gcn_w
    cvt_f32_f16<<<(int)(((long long)N * WIN + 255) / 256), 256, 0, stream>>>(h, hh, (long long)N * WIN);
    cvt_f32_f16<<<(int)(((long long)WIN * WOUT + 255) / 256), 256, 0, stream>>>(gcn_w, gw16, (long long)WIN * WOUT);
    dim3 ghw((WOUT + BN - 1) / BN, (N + BM - 1) / BM, 1);
    wmma_gemm<false,true,false,false,false><<<ghw, 256, 0, stream>>>(
        hh, gw16, hw, WOUT, WIN, WIN, WOUT, WOUT, 0, 0, 0, nullptr, nullptr, 0);

    // 7) S[c] = dout[c] * hw  (f16)
    scale_src<<<(int)(((long long)2 * N * WOUT + 255) / 256), 256, 0, stream>>>(hw, dout, S, N, WOUT);

    // 8) X[:, c*128:(c+1)*128] = relu(H[c]^T @ S[c] * din[c] + gcn_b)
    dim3 gagg((WOUT + BN - 1) / BN, (N + BM - 1) / BM, C);
    wmma_gemm<true,true,true,true,true><<<gagg, 256, 0, stream>>>(
        Hh, S, X, WOUT, N, N, WOUT, C * WOUT,
        NN, (long long)N * WOUT, (long long)WOUT, gcn_b, din, (long long)N);

    // 9) X1 = relu(X @ lin1_w + lin1_b)
    cvt_f32_f16<<<(int)(((long long)N * C * WOUT + 255) / 256), 256, 0, stream>>>(X, Xh, (long long)N * C * WOUT);
    cvt_f32_f16<<<(int)(((long long)C * WOUT * WOUT + 255) / 256), 256, 0, stream>>>(l1w, l1wh, (long long)C * WOUT * WOUT);
    wmma_gemm<false,true,true,false,true><<<ghw, 256, 0, stream>>>(
        Xh, l1wh, X1, WOUT, C * WOUT, C * WOUT, WOUT, WOUT,
        0, 0, 0, l1b, nullptr, 0);

    // 10) out = X1 @ lin2_w + lin2_b
    cvt_f32_f16<<<(int)(((long long)N * WOUT + 255) / 256), 256, 0, stream>>>(X1, X1h, (long long)N * WOUT);
    cvt_f32_f16<<<(int)(((long long)WOUT * NCLS + 255) / 256), 256, 0, stream>>>(l2w, l2wh, (long long)WOUT * NCLS);
    dim3 gcls((NCLS + BN - 1) / BN, (N + BM - 1) / BM, 1);
    wmma_gemm<false,false,true,false,false><<<gcls, 256, 0, stream>>>(
        X1h, l2wh, out, NCLS, WOUT, WOUT, NCLS, NCLS,
        0, 0, 0, l2b, nullptr, 0);
}
